// TransformerEncoderBlock_73710228734446
// MI455X (gfx1250) — matrix operand order repaired and checked by reading, not compiled, not run on hardware
//
#include <hip/hip_runtime.h>
#include <stdint.h>

// ---------------------------------------------------------------------------
// Problem constants (from the reference)
// ---------------------------------------------------------------------------
#define BATCH   8
#define SEQ     1024
#define CDIM    768
#define HEADS   12
#define HDIM    64
#define HIDDEN  3072
#define TOK     (BATCH * SEQ)        // 8192 rows
#define QKVDIM  (3 * CDIM)           // 2304

// GEMM tiling
#define BM   128
#define BN   128
#define BKK  32
#define LDT  40                      // padded LDS row stride, GEMM tiles (elements)
#define LDK  72                      // padded LDS row stride, attention (144B = 9*16B)

typedef __attribute__((ext_vector_type(4)))  int    v4i;
typedef __attribute__((ext_vector_type(8)))  float  v8f;
typedef __attribute__((ext_vector_type(16))) __bf16 v16bf;

union FragU { v16bf v; int   i[8]; };
union AccU  { v8f   v; float f[8]; };

// ---------------------------------------------------------------------------
// Async global->LDS (CDNA5 ASYNCcnt path), with safe fallback
// ---------------------------------------------------------------------------
#if defined(__has_builtin)
#if __has_builtin(__builtin_amdgcn_global_load_async_to_lds_b128) && \
    __has_builtin(__builtin_amdgcn_s_wait_asynccnt)
#define HAVE_ASYNC_LDS 1
#endif
#endif
#ifndef HAVE_ASYNC_LDS
#define HAVE_ASYNC_LDS 0
#endif

__device__ __forceinline__ void copy16_g2l(const void* g, void* l) {
#if HAVE_ASYNC_LDS
    __builtin_amdgcn_global_load_async_to_lds_b128(
        (__attribute__((address_space(1))) v4i*)(g),
        (__attribute__((address_space(3))) v4i*)(l), 0, 0);
#else
    *(uint4*)l = *(const uint4*)g;
#endif
}

__device__ __forceinline__ void wait_async_copies() {
#if HAVE_ASYNC_LDS
    __builtin_amdgcn_s_wait_asynccnt(0);
#endif
}

__device__ __forceinline__ uint16_t f2bf(float f) {
    uint32_t u = __float_as_uint(f);
    uint32_t r = u + 0x7FFFu + ((u >> 16) & 1u);   // round to nearest even
    return (uint16_t)(r >> 16);
}

// ---------------------------------------------------------------------------
// fp32 -> bf16 conversion (weights), grid-stride
// ---------------------------------------------------------------------------
__global__ void cvt_bf16_kernel(const float* __restrict__ in,
                                uint16_t* __restrict__ out, int n) {
    int stride = gridDim.x * blockDim.x;
    for (int i = blockIdx.x * blockDim.x + threadIdx.x; i < n; i += stride)
        out[i] = f2bf(in[i]);
}

// ---------------------------------------------------------------------------
// LayerNorm: one block per token row of 768, fp32 in -> bf16 out
// ---------------------------------------------------------------------------
__global__ __launch_bounds__(256)
void ln_kernel(const float* __restrict__ x, const float* __restrict__ g,
               const float* __restrict__ b, uint16_t* __restrict__ out) {
    __shared__ float ss[8];
    __shared__ float sq[8];
    const int row  = blockIdx.x;
    const int t    = threadIdx.x;
    const int lane = t & 31;
    const int wave = t >> 5;
    const float* xr = x + (size_t)row * CDIM;

    float v0 = xr[t], v1 = xr[t + 256], v2 = xr[t + 512];
    float s = v0 + v1 + v2;
    float q = v0 * v0 + v1 * v1 + v2 * v2;
#pragma unroll
    for (int off = 16; off > 0; off >>= 1) {
        s += __shfl_xor(s, off, 32);
        q += __shfl_xor(q, off, 32);
    }
    if (lane == 0) { ss[wave] = s; sq[wave] = q; }
    __syncthreads();
    s = 0.f; q = 0.f;
#pragma unroll
    for (int w = 0; w < 8; ++w) { s += ss[w]; q += sq[w]; }
    const float mu  = s * (1.0f / CDIM);
    const float var = q * (1.0f / CDIM) - mu * mu;
    const float r   = rsqrtf(var + 1e-5f);

    uint16_t* o = out + (size_t)row * CDIM;
    o[t]       = f2bf((v0 - mu) * r * g[t]       + b[t]);
    o[t + 256] = f2bf((v1 - mu) * r * g[t + 256] + b[t + 256]);
    o[t + 512] = f2bf((v2 - mu) * r * g[t + 512] + b[t + 512]);
}

// ---------------------------------------------------------------------------
// Tiled bf16 WMMA GEMM with double-buffered (async) LDS tiles:
//   out[M,N] = A[M,K] @ W[N,K]^T  (+ fused epilogue)
//  EPI 0: qkv scatter -> q (scaled 1/8), k, vT   (all bf16, per-head layouts)
//  EPI 1: outF = resid + acc + bias              (fp32)
//  EPI 2: out0 = bf16(gelu(acc + bias))
// ---------------------------------------------------------------------------
template <int EPI>
__global__ __launch_bounds__(256)
void gemm_bf16_kernel(const uint16_t* __restrict__ A,
                      const uint16_t* __restrict__ W,
                      const float*    __restrict__ bias,
                      const float*    __restrict__ resid,
                      float*          __restrict__ outF,
                      uint16_t*       __restrict__ out0,
                      uint16_t*       __restrict__ out1,
                      uint16_t*       __restrict__ out2,
                      int M, int N, int K) {
    __shared__ uint16_t As[2][BM * LDT];
    __shared__ uint16_t Bs[2][BN * LDT];

    const int tid  = threadIdx.x;
    const int lane = tid & 31;
    const int wave = tid >> 5;
    const int wm   = wave & 1;        // 2 wave rows   (64 rows each)
    const int wn   = wave >> 1;       // 4 wave cols   (32 cols each)
    const int bm0  = blockIdx.y * BM;
    const int bn0  = blockIdx.x * BN;
    const int half = lane >> 4;       // 0 | 1
    const int l16  = lane & 15;

    auto load_tiles = [&](int k0, int buf) {
#pragma unroll
        for (int i = 0; i < 2; ++i) {
            int u   = tid + i * 256;
            int row = u >> 2;
            int seg = u & 3;
            copy16_g2l(A + (size_t)(bm0 + row) * K + k0 + seg * 8,
                       &As[buf][row * LDT + seg * 8]);
            copy16_g2l(W + (size_t)(bn0 + row) * K + k0 + seg * 8,
                       &Bs[buf][row * LDT + seg * 8]);
        }
    };

    AccU acc[4][2];
#pragma unroll
    for (int i = 0; i < 4; ++i)
#pragma unroll
        for (int j = 0; j < 2; ++j) {
            v8f z = {0.f, 0.f, 0.f, 0.f, 0.f, 0.f, 0.f, 0.f};
            acc[i][j].v = z;
        }

    load_tiles(0, 0);
    wait_async_copies();
    __syncthreads();

    int cur = 0;
    for (int k0 = 0; k0 < K; k0 += BKK) {
        const int nxt = cur ^ 1;
        if (k0 + BKK < K) load_tiles(k0 + BKK, nxt);   // prefetch next tile

        const uint16_t* Asc = As[cur];
        const uint16_t* Bsc = Bs[cur];
        FragU fb[2];
#pragma unroll
        for (int j = 0; j < 2; ++j) {
            const int n  = wn * 32 + j * 16 + l16;
            const int kb = half * 16;
#pragma unroll
            for (int v = 0; v < 8; ++v)
                fb[j].i[v] = *(const int*)(Bsc + n * LDT + kb + 2 * v);
        }
#pragma unroll
        for (int i = 0; i < 4; ++i) {
            FragU fa;
            const int m  = wm * 64 + i * 16 + l16;
            const int kh = half * 16;
#pragma unroll
            for (int v = 0; v < 8; ++v)
                fa.i[v] = *(const int*)(Asc + m * LDT + kh + 2 * v);
#pragma unroll
            for (int j = 0; j < 2; ++j)
                acc[i][j].v = __builtin_amdgcn_wmma_f32_16x16x32_bf16(
                    false, fa.v, false, fb[j].v, (short)0, acc[i][j].v, false, false);
        }
        wait_async_copies();
        __syncthreads();
        cur = nxt;
    }

    // ---- epilogue ----
#pragma unroll
    for (int i = 0; i < 4; ++i)
#pragma unroll
        for (int j = 0; j < 2; ++j)
#pragma unroll
            for (int v = 0; v < 8; ++v) {
                const int gm = bm0 + wm * 64 + i * 16 + v + 8 * half;
                const int gn = bn0 + wn * 32 + j * 16 + l16;
                float c = acc[i][j].f[v] + bias[gn];
                if constexpr (EPI == 0) {
                    const int t   = gn / CDIM;          // uniform in wave
                    const int rem = gn - t * CDIM;
                    const int hd  = rem >> 6;           // head (uniform)
                    const int dh  = rem & 63;
                    const int bb  = gm >> 10;
                    const int n   = gm & 1023;
                    const size_t bh = (size_t)(bb * HEADS + hd);
                    if (t == 0)
                        out0[(bh * SEQ + n) * HDIM + dh] = f2bf(c * 0.125f);
                    else if (t == 1)
                        out1[(bh * SEQ + n) * HDIM + dh] = f2bf(c);
                    else
                        out2[(bh * HDIM + dh) * SEQ + n] = f2bf(c);
                } else if constexpr (EPI == 1) {
                    const size_t idx = (size_t)gm * N + gn;
                    outF[idx] = resid[idx] + c;
                } else {  // EPI == 2 : exact GELU
                    float ge = 0.5f * c * (1.0f + erff(c * 0.70710678118654752f));
                    out0[(size_t)gm * N + gn] = f2bf(ge);
                }
            }
}

// ---------------------------------------------------------------------------
// Flash-style attention: one block per (b, h, 64-query tile); 4 waves x 16 rows.
// 64-key chunks staged in double-buffered LDS (async copies); q pre-scaled 1/8.
// o[b][n][h*64+dh] written bf16 for the proj GEMM.
// ---------------------------------------------------------------------------
__global__ __launch_bounds__(128)
void attn_kernel(const uint16_t* __restrict__ qb,
                 const uint16_t* __restrict__ kb,
                 const uint16_t* __restrict__ vtb,
                 uint16_t* __restrict__ ob) {
    __shared__ uint16_t Ks[2][64 * LDK];     // [key][dh]
    __shared__ uint16_t Vs[2][64 * LDK];     // [dh][key]
    __shared__ uint16_t Pbuf[4 * 16 * LDK];  // per-wave P transpose staging

    const int tid  = threadIdx.x;
    const int lane = tid & 31;
    const int wave = tid >> 5;
    const int half = lane >> 4;
    const int l16  = lane & 15;

    const int blk = blockIdx.x;
    const int bh  = blk >> 4;            // 0..95
    const int qt  = blk & 15;            // 64-row query tile
    const int bb  = bh / HEADS;
    const int hd  = bh - bb * HEADS;

    const uint16_t* Q  = qb  + (size_t)bh * SEQ * HDIM;
    const uint16_t* Kp = kb  + (size_t)bh * SEQ * HDIM;
    const uint16_t* Vt = vtb + (size_t)bh * HDIM * SEQ;
    uint16_t* Pw = Pbuf + wave * 16 * LDK;

    const int m0 = qt * 64 + wave * 16;

    auto load_chunk = [&](int c, int buf) {
        // 512 uint4 per matrix chunk, 4 per thread each for K and V
#pragma unroll
        for (int i = 0; i < 4; ++i) {
            int u   = tid + i * 128;
            int row = u >> 3;            // 0..63
            int seg = u & 7;
            copy16_g2l(Kp + (size_t)(c + row) * HDIM + seg * 8,
                       &Ks[buf][row * LDK + seg * 8]);
            copy16_g2l(Vt + (size_t)row * SEQ + c + seg * 8,
                       &Vs[buf][row * LDK + seg * 8]);
        }
    };

    // Q fragments (A-layout), two K=32 steps over dh
    FragU qa[2];
    {
        const int m  = m0 + l16;
        const int kh = half * 16;
#pragma unroll
        for (int kk = 0; kk < 2; ++kk)
#pragma unroll
            for (int v = 0; v < 8; ++v) {
                const int dh = kk * 32 + kh + 2 * v;
                qa[kk].i[v] = *(const int*)(Q + (size_t)m * HDIM + dh);
            }
    }

    AccU o[4];
    float mrun[8], lrun[8];
#pragma unroll
    for (int j = 0; j < 4; ++j) {
        v8f z = {0.f, 0.f, 0.f, 0.f, 0.f, 0.f, 0.f, 0.f};
        o[j].v = z;
    }
#pragma unroll
    for (int v = 0; v < 8; ++v) { mrun[v] = -3.0e38f; lrun[v] = 0.f; }

    load_chunk(0, 0);
    wait_async_copies();
    __syncthreads();

    int cur = 0;
    for (int c = 0; c < SEQ; c += 64) {
        const int nxt = cur ^ 1;
        if (c + 64 < SEQ) load_chunk(c + 64, nxt);   // prefetch next K/V chunk
        const uint16_t* Ksc = Ks[cur];
        const uint16_t* Vsc = Vs[cur];

        // S = Q K^T : 4 key tiles x 2 dh k-steps
        AccU s[4];
#pragma unroll
        for (int j = 0; j < 4; ++j) {
            v8f z = {0.f, 0.f, 0.f, 0.f, 0.f, 0.f, 0.f, 0.f};
            s[j].v = z;
        }
#pragma unroll
        for (int kk = 0; kk < 2; ++kk)
#pragma unroll
            for (int j = 0; j < 4; ++j) {
                FragU fk;
                const int key = j * 16 + l16;
                const int dhb = kk * 32 + half * 16;
#pragma unroll
                for (int v = 0; v < 8; ++v)
                    fk.i[v] = *(const int*)(Ksc + key * LDK + dhb + 2 * v);
                s[j].v = __builtin_amdgcn_wmma_f32_16x16x32_bf16(
                    false, qa[kk].v, false, fk.v, (short)0, s[j].v, false, false);
            }

        // online softmax update (rows live across the 16-lane half)
#pragma unroll
        for (int v = 0; v < 8; ++v) {
            float cm = fmaxf(fmaxf(s[0].f[v], s[1].f[v]),
                             fmaxf(s[2].f[v], s[3].f[v]));
            cm = fmaxf(cm, __shfl_xor(cm, 1, 32));
            cm = fmaxf(cm, __shfl_xor(cm, 2, 32));
            cm = fmaxf(cm, __shfl_xor(cm, 4, 32));
            cm = fmaxf(cm, __shfl_xor(cm, 8, 32));
            const float mn    = fmaxf(mrun[v], cm);
            const float alpha = __expf(mrun[v] - mn);
            float p[4];
#pragma unroll
            for (int j = 0; j < 4; ++j) p[j] = __expf(s[j].f[v] - mn);
            float rs = (p[0] + p[1]) + (p[2] + p[3]);
            rs += __shfl_xor(rs, 1, 32);
            rs += __shfl_xor(rs, 2, 32);
            rs += __shfl_xor(rs, 4, 32);
            rs += __shfl_xor(rs, 8, 32);
            lrun[v] = lrun[v] * alpha + rs;
            mrun[v] = mn;
#pragma unroll
            for (int j = 0; j < 4; ++j) o[j].f[v] *= alpha;
            const int row = v + 8 * half;
#pragma unroll
            for (int j = 0; j < 4; ++j)
                Pw[row * LDK + j * 16 + l16] = f2bf(p[j]);
        }
        __builtin_amdgcn_wave_barrier();   // DS in-order per wave; block reordering

        // P as two A-fragments (16 x 64 over keys)
        FragU pa[2];
        {
            const int kh = half * 16;
#pragma unroll
            for (int a = 0; a < 2; ++a)
#pragma unroll
                for (int v = 0; v < 8; ++v)
                    pa[a].i[v] = *(const int*)(
                        Pw + l16 * LDK + a * 32 + kh + 2 * v);
        }
        // O += P @ V  (Vs layout [dh][key]: key pairs contiguous)
#pragma unroll
        for (int a = 0; a < 2; ++a)
#pragma unroll
            for (int j = 0; j < 4; ++j) {
                FragU fv;
                const int dh  = j * 16 + l16;
                const int kbp = a * 32 + half * 16;
#pragma unroll
                for (int v = 0; v < 8; ++v)
                    fv.i[v] = *(const int*)(Vsc + dh * LDK + kbp + 2 * v);
                o[j].v = __builtin_amdgcn_wmma_f32_16x16x32_bf16(
                    false, pa[a].v, false, fv.v, (short)0, o[j].v, false, false);
            }
        __builtin_amdgcn_wave_barrier();

        wait_async_copies();
        __syncthreads();
        cur = nxt;
    }

    // finalize: divide by row sums, write o[b][n][h*64+dh] bf16
#pragma unroll
    for (int v = 0; v < 8; ++v) {
        const float inv = 1.0f / lrun[v];
        const int   n   = m0 + v + 8 * half;
        const size_t rowb = ((size_t)bb * SEQ + n) * CDIM + hd * HDIM;
#pragma unroll
        for (int j = 0; j < 4; ++j)
            ob[rowb + j * 16 + l16] = f2bf(o[j].f[v] * inv);
    }
}

// ---------------------------------------------------------------------------
// Host launcher
// ---------------------------------------------------------------------------
extern "C" void kernel_launch(void* const* d_in, const int* in_sizes, int n_in,
                              void* d_out, int out_size, void* d_ws, size_t ws_size,
                              hipStream_t stream) {
    (void)in_sizes; (void)n_in; (void)out_size; (void)ws_size;
    const float* x      = (const float*)d_in[0];
    const float* n1g    = (const float*)d_in[1];
    const float* n1b    = (const float*)d_in[2];
    const float* qkv_w  = (const float*)d_in[3];
    const float* qkv_b  = (const float*)d_in[4];
    const float* proj_w = (const float*)d_in[5];
    const float* proj_b = (const float*)d_in[6];
    const float* n2g    = (const float*)d_in[7];
    const float* n2b    = (const float*)d_in[8];
    const float* fc1_w  = (const float*)d_in[9];
    const float* fc1_b  = (const float*)d_in[10];
    const float* fc2_w  = (const float*)d_in[11];
    const float* fc2_b  = (const float*)d_in[12];
    float* out = (float*)d_out;

    char* ws = (char*)d_ws;
    size_t off = 0;
    auto take = [&](size_t bytes) {
        char* p = ws + off;
        off = (off + bytes + 255) & ~(size_t)255;
        return p;
    };
    uint16_t* h1   = (uint16_t*)take((size_t)TOK * CDIM * 2);
    uint16_t* h2   = (uint16_t*)take((size_t)TOK * CDIM * 2);
    uint16_t* wqkv = (uint16_t*)take((size_t)QKVDIM * CDIM * 2);
    uint16_t* wprj = (uint16_t*)take((size_t)CDIM * CDIM * 2);
    uint16_t* wfc1 = (uint16_t*)take((size_t)HIDDEN * CDIM * 2);
    uint16_t* wfc2 = (uint16_t*)take((size_t)CDIM * HIDDEN * 2);
    uint16_t* qbuf = (uint16_t*)take((size_t)TOK * CDIM * 2);
    uint16_t* kbuf = (uint16_t*)take((size_t)TOK * CDIM * 2);
    uint16_t* vtbf = (uint16_t*)take((size_t)TOK * CDIM * 2);
    uint16_t* obuf = (uint16_t*)take((size_t)TOK * CDIM * 2);
    uint16_t* gbuf = (uint16_t*)take((size_t)TOK * HIDDEN * 2);
    float*    x2   = (float*)   take((size_t)TOK * CDIM * 4);

    // weights -> bf16
    cvt_bf16_kernel<<<dim3(512), 256, 0, stream>>>(qkv_w, wqkv, QKVDIM * CDIM);
    cvt_bf16_kernel<<<dim3(512), 256, 0, stream>>>(proj_w, wprj, CDIM * CDIM);
    cvt_bf16_kernel<<<dim3(512), 256, 0, stream>>>(fc1_w, wfc1, HIDDEN * CDIM);
    cvt_bf16_kernel<<<dim3(512), 256, 0, stream>>>(fc2_w, wfc2, CDIM * HIDDEN);

    // LN1 -> h1
    ln_kernel<<<dim3(TOK), 256, 0, stream>>>(x, n1g, n1b, h1);

    // QKV GEMM with scatter to q/k/vT
    gemm_bf16_kernel<0><<<dim3(QKVDIM / BN, TOK / BM), 256, 0, stream>>>(
        h1, wqkv, qkv_b, nullptr, nullptr, qbuf, kbuf, vtbf, TOK, QKVDIM, CDIM);

    // attention -> obuf
    attn_kernel<<<dim3(BATCH * HEADS * (SEQ / 64)), 128, 0, stream>>>(
        qbuf, kbuf, vtbf, obuf);

    // proj GEMM + residual (x) -> x2 (fp32)
    gemm_bf16_kernel<1><<<dim3(CDIM / BN, TOK / BM), 256, 0, stream>>>(
        obuf, wprj, proj_b, x, x2, nullptr, nullptr, nullptr, TOK, CDIM, CDIM);

    // LN2 -> h2
    ln_kernel<<<dim3(TOK), 256, 0, stream>>>(x2, n2g, n2b, h2);

    // FC1 + exact GELU -> gbuf (bf16)
    gemm_bf16_kernel<2><<<dim3(HIDDEN / BN, TOK / BM), 256, 0, stream>>>(
        h2, wfc1, fc1_b, nullptr, nullptr, gbuf, nullptr, nullptr, TOK, HIDDEN, CDIM);

    // FC2 + residual (x2) -> d_out (fp32)
    gemm_bf16_kernel<1><<<dim3(CDIM / BN, TOK / BM), 256, 0, stream>>>(
        gbuf, wfc2, fc2_b, x2, out, nullptr, nullptr, nullptr, TOK, CDIM, HIDDEN);
}
